// PCILTConv2d_25254407701007
// MI455X (gfx1250) — 8-bit matrix operand order repaired and checked by reading, compiled, not run on hardware
//
#include <hip/hip_runtime.h>
#include <stdint.h>

typedef int v8i __attribute__((ext_vector_type(8)));
typedef int v4i __attribute__((ext_vector_type(4)));

#define C_IN   128
#define H_IN   56
#define W_IN   56
#define HP     58
#define WP     58
#define O_OUT  256
#define KTOT   1152
#define NKC    18
#define HW     3136
#define NTOT   50176

#define XQ_BYTES (16 * HP * WP * C_IN)
#define QW_OFF   ((XQ_BYTES + 255) & ~255)
#define QW_BYTES (O_OUT * KTOT)

__global__ void zero_u32_kernel(uint32_t* __restrict__ p, int n32) {
    int i = blockIdx.x * blockDim.x + threadIdx.x;
    if (i < n32) p[i] = 0u;
}

__global__ void quantize_pad_kernel(const float* __restrict__ x, uint8_t* __restrict__ xq) {
    int i = blockIdx.x * blockDim.x + threadIdx.x;
    if (i >= 16 * C_IN * HW) return;
    int w = i % W_IN;
    int t = i / W_IN;
    int h = t % H_IN;  t /= H_IN;
    int c = t % C_IN;
    int b = t / C_IN;
    float q = fminf(fmaxf(__builtin_rintf(x[i] * 255.0f), 0.0f), 255.0f);
    xq[((size_t)(b * HP + h + 1) * WP + (w + 1)) * C_IN + c] = (uint8_t)q;
}

__global__ void extract_qw_kernel(const float* __restrict__ pcilt, uint8_t* __restrict__ qwS) {
    int t = blockIdx.x * blockDim.x + threadIdx.x;
    if (t >= O_OUT * KTOT) return;
    int j    = t & 31;
    int idx  = t >> 5;
    int lane = idx & 31;
    idx    >>= 5;
    int kc   = idx % NKC;
    int t_o  = idx / NKC;
    int laneLo = lane & 15;
    int grp    = lane >> 4;
    int o  = t_o * 16 + laneLo;
    int kl = ((j >> 4) << 5) + grp * 16 + (j & 15);
    int k  = kc * 64 + kl;
    int khkw = k >> 7;
    int c    = k & 127;
    int kh   = khkw / 3;
    int kw   = khkw - kh * 3;
    size_t src = ((((size_t)o * C_IN + c) * 3 + kh) * 3 + kw) * 256 + 1;
    qwS[t] = (uint8_t)__builtin_rintf(pcilt[src]);
}

__global__ __launch_bounds__(128, 1)
void pcilt_conv_wmma_kernel(const uint8_t* __restrict__ xq,
                            const uint8_t* __restrict__ qwS,
                            const float*   __restrict__ bias,
                            float*         __restrict__ out) {
    const int lane   = threadIdx.x & 31;
    const int wave   = threadIdx.x >> 5;
    const int laneLo = lane & 15;
    const int grp    = lane >> 4;

    const int n0   = blockIdx.x * 32 + laneLo;
    const int n1   = n0 + 16;
    const int b0   = n0 / HW;
    const int r0   = n0 - b0 * HW;
    const int h0   = r0 / W_IN;
    const int w0   = r0 - h0 * W_IN;
    const int b1   = n1 / HW;
    const int r1   = n1 - b1 * HW;
    const int h1   = r1 / W_IN;
    const int w1   = r1 - h1 * W_IN;
    const uint8_t* colBase0 = xq + ((size_t)(b0 * HP + h0) * WP + w0) * C_IN;
    const uint8_t* colBase1 = xq + ((size_t)(b1 * HP + h1) * WP + w1) * C_IN;

    const int oBase = (blockIdx.y * 4 + wave) * 32;
    const int tO    = oBase >> 4;
    const uint8_t* aPtr0 = qwS + ((size_t)tO * NKC * 32 + lane) * 32;
    const uint8_t* aPtr1 = aPtr0 + (size_t)NKC * 1024;

    v8i acc00 = {};
    v8i acc01 = {};
    v8i acc10 = {};
    v8i acc11 = {};

#pragma unroll
    for (int kc = 0; kc < NKC; ++kc) {
        const int khkw = kc >> 1;
        const int kh   = khkw / 3;
        const int kw   = khkw - kh * 3;
        const int bOff = (kh * WP + kw) * C_IN + ((kc & 1) << 6);
        const int aOff = kc * 1024;

        v4i a0lo = *(const v4i*)(aPtr0 + aOff);
        v4i a0hi = *(const v4i*)(aPtr0 + aOff + 16);
        v4i a1lo = *(const v4i*)(aPtr1 + aOff);
        v4i a1hi = *(const v4i*)(aPtr1 + aOff + 16);
        v8i aF0; aF0[0]=a0lo[0]; aF0[1]=a0lo[1]; aF0[2]=a0lo[2]; aF0[3]=a0lo[3];
                 aF0[4]=a0hi[0]; aF0[5]=a0hi[1]; aF0[6]=a0hi[2]; aF0[7]=a0hi[3];
        v8i aF1; aF1[0]=a1lo[0]; aF1[1]=a1lo[1]; aF1[2]=a1lo[2]; aF1[3]=a1lo[3];
                 aF1[4]=a1hi[0]; aF1[5]=a1hi[1]; aF1[6]=a1hi[2]; aF1[7]=a1hi[3];

        const uint8_t* bp0 = colBase0 + bOff + grp * 16;
        const uint8_t* bp1 = colBase1 + bOff + grp * 16;
        v4i b00 = *(const v4i*)(bp0);
        v4i b01 = *(const v4i*)(bp0 + 32);
        v4i b10 = *(const v4i*)(bp1);
        v4i b11 = *(const v4i*)(bp1 + 32);
        v8i bF0; bF0[0]=b00[0]; bF0[1]=b00[1]; bF0[2]=b00[2]; bF0[3]=b00[3];
                 bF0[4]=b01[0]; bF0[5]=b01[1]; bF0[6]=b01[2]; bF0[7]=b01[3];
        v8i bF1; bF1[0]=b10[0]; bF1[1]=b10[1]; bF1[2]=b10[2]; bF1[3]=b10[3];
                 bF1[4]=b11[0]; bF1[5]=b11[1]; bF1[6]=b11[2]; bF1[7]=b11[3];

        acc00 = __builtin_amdgcn_wmma_i32_16x16x64_iu8(false, aF0, false, bF0, acc00, false, false);
        acc01 = __builtin_amdgcn_wmma_i32_16x16x64_iu8(false, aF0, false, bF1, acc01, false, false);
        acc10 = __builtin_amdgcn_wmma_i32_16x16x64_iu8(false, aF1, false, bF0, acc10, false, false);
        acc11 = __builtin_amdgcn_wmma_i32_16x16x64_iu8(false, aF1, false, bF1, acc11, false, false);
    }

    float* ob0 = out + (size_t)b0 * O_OUT * HW + (size_t)h0 * W_IN + w0;
    float* ob1 = out + (size_t)b1 * O_OUT * HW + (size_t)h1 * W_IN + w1;
#pragma unroll
    for (int r = 0; r < 8; ++r) {
        const int m0 = oBase + r + grp * 8;
        const int m1 = m0 + 16;
        const float bz0 = bias[m0];
        const float bz1 = bias[m1];
        ob0[(size_t)m0 * HW] = (float)acc00[r] + bz0;
        ob1[(size_t)m0 * HW] = (float)acc01[r] + bz0;
        ob0[(size_t)m1 * HW] = (float)acc10[r] + bz1;
        ob1[(size_t)m1 * HW] = (float)acc11[r] + bz1;
    }
}

extern "C" void kernel_launch(void* const* d_in, const int* in_sizes, int n_in,
                              void* d_out, int out_size, void* d_ws, size_t ws_size,
                              hipStream_t stream) {
    const float* x     = (const float*)d_in[0];
    const float* pcilt = (const float*)d_in[1];
    const float* bias  = (const float*)d_in[2];
    float*       out   = (float*)d_out;

    uint8_t* ws  = (uint8_t*)d_ws;
    uint8_t* xq  = ws;
    uint8_t* qwS = ws + QW_OFF;

    const int n32 = XQ_BYTES / 4;
    zero_u32_kernel<<<(n32 + 255) / 256, 256, 0, stream>>>((uint32_t*)xq, n32);

    const int nx = 16 * C_IN * HW;
    quantize_pad_kernel<<<(nx + 255) / 256, 256, 0, stream>>>(x, xq);

    const int nq = O_OUT * KTOT;
    extract_qw_kernel<<<(nq + 255) / 256, 256, 0, stream>>>(pcilt, qwS);

    dim3 grid(NTOT / 32, 2);
    pcilt_conv_wmma_kernel<<<grid, 128, 0, stream>>>(xq, qwS, bias, out);
}
